// GAT_13039520710886
// MI455X (gfx1250) — matrix operand order repaired and checked by reading, not compiled, not run on hardware
//
#include <hip/hip_runtime.h>
#include <hip/hip_bf16.h>

// ---------------------------------------------------------------------------
// GAT forward for MI455X (gfx1250, wave32). GEMMs via v_wmma_f32_16x16x32_f16.
// ---------------------------------------------------------------------------

typedef _Float16 v16h __attribute__((ext_vector_type(16)));
typedef _Float16 v8h  __attribute__((ext_vector_type(8)));
typedef float    v8f  __attribute__((ext_vector_type(8)));

#define OUTC 256
#define HEADS 8
#define HD 32

// ---- ordered-uint float max helpers (monotonic bijection f32 -> u32) -------
__device__ __forceinline__ unsigned fflip(float f) {
    unsigned u = __float_as_uint(f);
    return (u & 0x80000000u) ? ~u : (u | 0x80000000u);
}
__device__ __forceinline__ float funflip(unsigned u) {
    return __uint_as_float((u & 0x80000000u) ? (u & 0x7FFFFFFFu) : ~u);
}
#define NEG_INF_FLIPPED 0x007FFFFFu   // fflip(-inf)

__device__ __forceinline__ float lrelu(float x, float s) { return x > 0.0f ? x : s * x; }

// combine two 8-half runs into a 16-half WMMA fragment register set
__device__ __forceinline__ v16h frag2(v8h lo, v8h hi) {
    return __builtin_shufflevector(lo, hi, 0, 1, 2, 3, 4, 5, 6, 7,
                                           8, 9, 10, 11, 12, 13, 14, 15);
}

// ---------------------------------------------------------------------------
// Conversions
// ---------------------------------------------------------------------------
__global__ void k_f32_to_f16(const float* __restrict__ in, _Float16* __restrict__ out, long n) {
    long i = (long)blockIdx.x * blockDim.x + threadIdx.x;
    if (i < n) out[i] = (_Float16)in[i];
}

// W[h,k,d] (f32, [8,128,32]) -> w16[(h*32+d)*128 + k]  (Bt layout: K-contiguous)
__global__ void k_convert_W(const float* __restrict__ W, _Float16* __restrict__ w16) {
    int i = blockIdx.x * blockDim.x + threadIdx.x;   // over 8*128*32 = 32768
    if (i >= HEADS * 128 * HD) return;
    int h = i >> 12;            // /4096
    int r = i & 4095;
    int k = r >> 5;             // /32
    int d = r & 31;
    w16[(long)((h << 5) + d) * 128 + k] = (_Float16)W[i];
}

// ---------------------------------------------------------------------------
// WMMA GEMM: C[M,Nc] = A[M,K](f16,row-major) * Bt[Nc,K](f16,row-major)^T + bias
// One wave computes a 16x64 strip of C (4 accumulators). Register double-
// buffering: next K-step fragments are loaded before the current WMMAs issue,
// overlapping vmem with the matrix pipe. K % 32 == 0, Nc % 64 == 0.
// Fragment layouts per CDNA5 ISA 7.12.2:
//   A (16-bit 16x32): lane half h holds K runs [h*8, h*8+7] and [16+h*8, 16+h*8+7]
//   B (16-bit 32x16): lane half h holds K runs [h*8, h*8+7] and [16+h*8, 16+h*8+7], col = lane&15
//   C (f32 16x16):    VGPR v -> row = half*8 + v, col = lane&15
// ---------------------------------------------------------------------------
__global__ void k_wmma_gemm(const _Float16* __restrict__ A,
                            const _Float16* __restrict__ Bt,
                            const float* __restrict__ bias,
                            float* __restrict__ C,
                            int M, int K, int Nc)
{
    const int wavesPerBlock = blockDim.x >> 5;
    const int wave = threadIdx.x >> 5;
    const int lane = threadIdx.x & 31;
    const int tilesM  = (M + 15) >> 4;
    const int groupsN = Nc >> 6;                       // 64-wide column groups
    long job = (long)blockIdx.x * wavesPerBlock + wave;
    if (job >= (long)tilesM * groupsN) return;         // wave-uniform exit
    const int tm = (int)(job / groupsN);
    const int gn = (int)(job % groupsN);
    const int row0 = tm << 4;
    const int col0 = gn << 6;
    const int half = lane >> 4;                        // 0/1 lane group
    const int l15  = lane & 15;

    int arow = row0 + l15;
    if (arow >= M) arow = M - 1;                       // branch-free tail clamp
    const _Float16* __restrict__ Ap = A + (long)arow * K + half * 8;
    const _Float16* __restrict__ Bp = Bt + (long)(col0 + l15) * K + half * 8;
    const long bcs = (long)16 * K;                     // 16-column tile stride in Bt

    v8f acc[4] = {v8f{}, v8f{}, v8f{}, v8f{}};

    // prologue: first fragment set
    v16h a = frag2(*(const v8h*)Ap, *(const v8h*)(Ap + 16));
    v16h b[4];
#pragma unroll
    for (int t = 0; t < 4; ++t) {
        const _Float16* bp = Bp + t * bcs;
        b[t] = frag2(*(const v8h*)bp, *(const v8h*)(bp + 16));
    }

    // steady state: issue next-step loads, then WMMAs on current fragments
#pragma unroll 2
    for (int k0 = 0; k0 + 32 < K; k0 += 32) {
        const int kn = k0 + 32;
        v16h an = frag2(*(const v8h*)(Ap + kn), *(const v8h*)(Ap + kn + 16));
        v16h bn[4];
#pragma unroll
        for (int t = 0; t < 4; ++t) {
            const _Float16* bp = Bp + t * bcs + kn;
            bn[t] = frag2(*(const v8h*)bp, *(const v8h*)(bp + 16));
        }
#pragma unroll
        for (int t = 0; t < 4; ++t)
            acc[t] = __builtin_amdgcn_wmma_f32_16x16x32_f16(false, a, false, b[t],
                                                            (short)0, acc[t], false, false);
        a = an;
#pragma unroll
        for (int t = 0; t < 4; ++t) b[t] = bn[t];
    }
    // epilogue: last K-step
#pragma unroll
    for (int t = 0; t < 4; ++t)
        acc[t] = __builtin_amdgcn_wmma_f32_16x16x32_f16(false, a, false, b[t],
                                                        (short)0, acc[t], false, false);

    const int crow = row0 + half * 8;
    const bool full = (row0 + 16 <= M);                // wave-uniform
#pragma unroll
    for (int t = 0; t < 4; ++t) {
        const int ccol = col0 + t * 16 + l15;
        const float bv = bias ? bias[ccol] : 0.0f;
        if (full) {
#pragma unroll
            for (int v = 0; v < 8; ++v)
                C[(long)(crow + v) * Nc + ccol] = acc[t][v] + bv;
        } else {
#pragma unroll
            for (int v = 0; v < 8; ++v)
                if (crow + v < M) C[(long)(crow + v) * Nc + ccol] = acc[t][v] + bv;
        }
    }
}

// ---------------------------------------------------------------------------
// Per-node attention coefficients: as[n,h] = <xp[n,h,:], a_src[h,:]>, same dst
// ---------------------------------------------------------------------------
__global__ void k_alpha(const float* __restrict__ xp, const float* __restrict__ a_src,
                        const float* __restrict__ a_dst, float* __restrict__ as,
                        float* __restrict__ ad, int N)
{
    int i = blockIdx.x * blockDim.x + threadIdx.x;    // over N*H
    if (i >= N * HEADS) return;
    int n = i >> 3, h = i & 7;
    const float* v = xp + (long)n * OUTC + h * HD;
    const float* s = a_src + h * HD;
    const float* d = a_dst + h * HD;
    float accs = 0.0f, accd = 0.0f;
#pragma unroll
    for (int t = 0; t < HD; ++t) { float x = v[t]; accs += x * s[t]; accd += x * d[t]; }
    as[i] = accs;
    ad[i] = accd;
}

// ---------------------------------------------------------------------------
// Init kernels
// ---------------------------------------------------------------------------
__global__ void k_init_agg(float* __restrict__ agg, const float* __restrict__ conv_b, long n) {
    long i = (long)blockIdx.x * blockDim.x + threadIdx.x;
    if (i < n) agg[i] = conv_b[i & (OUTC - 1)];
}
__global__ void k_init_seg(unsigned* __restrict__ segmax, float* __restrict__ segsum, int n) {
    int i = blockIdx.x * blockDim.x + threadIdx.x;
    if (i < n) { segmax[i] = NEG_INF_FLIPPED; segsum[i] = 0.0f; }
}
__global__ void k_init_pool(float* __restrict__ xg, unsigned* __restrict__ gmax,
                            float* __restrict__ gsum) {
    int i = threadIdx.x;
    if (i < OUTC) xg[i] = 0.0f;
    if (i == 0) { *gmax = NEG_INF_FLIPPED; *gsum = 0.0f; }
}

// ---------------------------------------------------------------------------
// Edge passes (Etot = E real edges + N self loops)
// ---------------------------------------------------------------------------
__global__ void k_edge_max(const int* __restrict__ ei, long E, int N,
                           const float* __restrict__ as, const float* __restrict__ ad,
                           unsigned* __restrict__ segmax)
{
    long e = (long)blockIdx.x * blockDim.x + threadIdx.x;
    if (e >= E + N) return;
    int s = (e < E) ? ei[e]     : (int)(e - E);
    int d = (e < E) ? ei[E + e] : (int)(e - E);
#pragma unroll
    for (int h = 0; h < HEADS; ++h) {
        float l = lrelu(as[s * HEADS + h] + ad[d * HEADS + h], 0.2f);
        atomicMax(&segmax[d * HEADS + h], fflip(l));
    }
}

__global__ void k_edge_sum(const int* __restrict__ ei, long E, int N,
                           const float* __restrict__ as, const float* __restrict__ ad,
                           const unsigned* __restrict__ segmax, float* __restrict__ segsum)
{
    long e = (long)blockIdx.x * blockDim.x + threadIdx.x;
    if (e >= E + N) return;
    int s = (e < E) ? ei[e]     : (int)(e - E);
    int d = (e < E) ? ei[E + e] : (int)(e - E);
#pragma unroll
    for (int h = 0; h < HEADS; ++h) {
        float l = lrelu(as[s * HEADS + h] + ad[d * HEADS + h], 0.2f);
        float m = funflip(segmax[d * HEADS + h]);
        atomicAdd(&segsum[d * HEADS + h], __expf(l - m));
    }
}

// one thread per (edge, head): gather 32 f32 of xp[src], scatter-add into agg[dst]
__global__ void k_edge_scatter(const int* __restrict__ ei, long E, int N,
                               const float* __restrict__ as, const float* __restrict__ ad,
                               const unsigned* __restrict__ segmax,
                               const float* __restrict__ segsum,
                               const float* __restrict__ xp, float* __restrict__ agg)
{
    long t = (long)blockIdx.x * blockDim.x + threadIdx.x;
    if (t >= (E + N) * HEADS) return;
    long e = t >> 3;
    int h = (int)(t & 7);
    int s = (e < E) ? ei[e]     : (int)(e - E);
    int d = (e < E) ? ei[E + e] : (int)(e - E);
    float l = lrelu(as[s * HEADS + h] + ad[d * HEADS + h], 0.2f);
    float m = funflip(segmax[d * HEADS + h]);
    float alpha = __expf(l - m) / (segsum[d * HEADS + h] + 1e-16f);
    const float* src = xp  + (long)s * OUTC + h * HD;
    float*       dst = agg + (long)d * OUTC + h * HD;
    __builtin_prefetch(src, 0, 0);                    // global_prefetch_b8
#pragma unroll
    for (int c = 0; c < HD; ++c) atomicAdd(&dst[c], alpha * src[c]);
}

// ---------------------------------------------------------------------------
// sa = softmax(leaky_relu(y1,0.01)); x2 = leaky_relu(x_local*sa,0.2) -> f16
// one wave per node, 8 channels per lane
// ---------------------------------------------------------------------------
__global__ void k_safuse(const float* __restrict__ xl, const float* __restrict__ y1,
                         _Float16* __restrict__ x2_16, int N)
{
    int n = blockIdx.x * (blockDim.x >> 5) + (threadIdx.x >> 5);
    if (n >= N) return;
    int lane = threadIdx.x & 31;
    float v[8];
    float mx = -3.4e38f;
#pragma unroll
    for (int j = 0; j < 8; ++j) {
        float y = lrelu(y1[(long)n * OUTC + j * 32 + lane], 0.01f);
        v[j] = y;
        mx = fmaxf(mx, y);
    }
#pragma unroll
    for (int o = 16; o > 0; o >>= 1) mx = fmaxf(mx, __shfl_xor(mx, o, 32));
    float sm = 0.0f;
#pragma unroll
    for (int j = 0; j < 8; ++j) { v[j] = __expf(v[j] - mx); sm += v[j]; }
#pragma unroll
    for (int o = 16; o > 0; o >>= 1) sm += __shfl_xor(sm, o, 32);
    float rs = 1.0f / sm;
#pragma unroll
    for (int j = 0; j < 8; ++j) {
        long idx = (long)n * OUTC + j * 32 + lane;
        float x = lrelu(xl[idx] * v[j] * rs, 0.2f);
        x2_16[idx] = (_Float16)x;
    }
}

// ---------------------------------------------------------------------------
// LayerNorm + L2 normalize + gate logit. one wave per node. xfin -> d_out
// ---------------------------------------------------------------------------
__global__ void k_ln_l2_gate(const float* __restrict__ y2, const float* __restrict__ ln_w,
                             const float* __restrict__ ln_b, const float* __restrict__ gate_w,
                             const float* __restrict__ gate_b, float* __restrict__ xfin,
                             float* __restrict__ gate, int N)
{
    int n = blockIdx.x * (blockDim.x >> 5) + (threadIdx.x >> 5);
    if (n >= N) return;
    int lane = threadIdx.x & 31;
    float v[8];
    float s = 0.0f;
#pragma unroll
    for (int j = 0; j < 8; ++j) { v[j] = y2[(long)n * OUTC + j * 32 + lane]; s += v[j]; }
#pragma unroll
    for (int o = 16; o > 0; o >>= 1) s += __shfl_xor(s, o, 32);
    float mu = s * (1.0f / OUTC);
    float vs = 0.0f;
#pragma unroll
    for (int j = 0; j < 8; ++j) { float d = v[j] - mu; vs += d * d; }
#pragma unroll
    for (int o = 16; o > 0; o >>= 1) vs += __shfl_xor(vs, o, 32);
    float inv = rsqrtf(vs * (1.0f / OUTC) + 1e-5f);
    float l2 = 0.0f;
#pragma unroll
    for (int j = 0; j < 8; ++j) {
        int c = j * 32 + lane;
        v[j] = (v[j] - mu) * inv * ln_w[c] + ln_b[c];
        l2 += v[j] * v[j];
    }
#pragma unroll
    for (int o = 16; o > 0; o >>= 1) l2 += __shfl_xor(l2, o, 32);
    float rinv = 1.0f / fmaxf(sqrtf(l2), 1e-12f);
    float g = 0.0f;
#pragma unroll
    for (int j = 0; j < 8; ++j) {
        int c = j * 32 + lane;
        float x = v[j] * rinv;
        xfin[(long)n * OUTC + c] = x;
        g += x * gate_w[c];
    }
#pragma unroll
    for (int o = 16; o > 0; o >>= 1) g += __shfl_xor(g, o, 32);
    if (lane == 0) gate[n] = g + gate_b[0];
}

// ---------------------------------------------------------------------------
// Global softmax pooling over all nodes (batch is all-zero, one graph)
// ---------------------------------------------------------------------------
__global__ void k_gate_max(const float* __restrict__ gate, unsigned* __restrict__ gmax, int N) {
    int i = blockIdx.x * blockDim.x + threadIdx.x;
    if (i < N) atomicMax(gmax, fflip(gate[i]));
}
__global__ void k_gate_sum(const float* __restrict__ gate, const unsigned* __restrict__ gmax,
                           float* __restrict__ gsum, int N) {
    int i = blockIdx.x * blockDim.x + threadIdx.x;
    if (i < N) atomicAdd(gsum, __expf(gate[i] - funflip(*gmax)));
}
__global__ void k_pool(const float* __restrict__ xfin, const float* __restrict__ gate,
                       const unsigned* __restrict__ gmax, const float* __restrict__ gsum,
                       float* __restrict__ xg, int N)
{
    long t = (long)blockIdx.x * blockDim.x + threadIdx.x;
    if (t >= (long)N * 32) return;
    int n = (int)(t >> 5), lane = (int)(t & 31);
    float g = __expf(gate[n] - funflip(*gmax)) / (*gsum + 1e-16f);
#pragma unroll
    for (int j = 0; j < 8; ++j) {
        int c = j * 32 + lane;
        atomicAdd(&xg[c], g * xfin[(long)n * OUTC + c]);
    }
}

// ga = softmax(relu(xg @ gfc_w^T + gfc_b)); single 256-thread block
__global__ void k_ga(const float* __restrict__ xg, const float* __restrict__ gfc_w,
                     const float* __restrict__ gfc_b, float* __restrict__ ga)
{
    __shared__ float red[OUTC];
    int j = threadIdx.x;
    float z = gfc_b[j];
    for (int k = 0; k < OUTC; ++k) z += xg[k] * gfc_w[j * OUTC + k];
    z = fmaxf(z, 0.0f);
    red[j] = z; __syncthreads();
    for (int s = 128; s > 0; s >>= 1) { if (j < s) red[j] = fmaxf(red[j], red[j + s]); __syncthreads(); }
    float mx = red[0]; __syncthreads();
    float e = __expf(z - mx);
    red[j] = e; __syncthreads();
    for (int s = 128; s > 0; s >>= 1) { if (j < s) red[j] += red[j + s]; __syncthreads(); }
    ga[j] = e / red[0];
}

__global__ void k_final(float* __restrict__ out, const float* __restrict__ ga, long n) {
    long i = (long)blockIdx.x * blockDim.x + threadIdx.x;
    if (i < n) out[i] *= ga[i & (OUTC - 1)];
}

// ---------------------------------------------------------------------------
// Host launcher
// ---------------------------------------------------------------------------
extern "C" void kernel_launch(void* const* d_in, const int* in_sizes, int n_in,
                              void* d_out, int out_size, void* d_ws, size_t ws_size,
                              hipStream_t stream)
{
    (void)n_in; (void)out_size; (void)ws_size;
    const float* x       = (const float*)d_in[0];
    const int*   ei      = (const int*)  d_in[1];
    // d_in[2] = edge_attr (ignored: edge_dim=None), d_in[3] = batch (all zero)
    const float* W       = (const float*)d_in[4];
    const float* a_src   = (const float*)d_in[5];
    const float* a_dst   = (const float*)d_in[6];
    const float* conv_b  = (const float*)d_in[7];
    const float* fc_w    = (const float*)d_in[8];
    const float* fc_b    = (const float*)d_in[9];
    const float* ln_w    = (const float*)d_in[10];
    const float* ln_b    = (const float*)d_in[11];
    const float* gate_w  = (const float*)d_in[12];
    const float* gate_b  = (const float*)d_in[13];
    const float* gfc_w   = (const float*)d_in[14];
    const float* gfc_b   = (const float*)d_in[15];

    const int  N    = in_sizes[0] / 128;
    const long E    = in_sizes[1] / 2;
    const long Etot = E + N;

    // ---- workspace carve-up (all 16B-aligned for b128 fragment loads) ----
    float*    xp     = (float*)d_ws;                    // [N,256] xp, later y2
    float*    agg    = xp  + (long)N * OUTC;            // [N,256] x_local
    float*    y1     = agg + (long)N * OUTC;            // [N,256]
    _Float16* h16    = (_Float16*)(y1 + (long)N * OUTC);// [N,256] f16 (x, then x2)
    _Float16* wf16   = h16 + (long)N * OUTC;            // 65536 halves (W-Bt / fc16)
    float*    as     = (float*)(wf16 + 65536);          // [N,8]
    float*    ad     = as + (long)N * HEADS;            // [N,8]
    unsigned* segmax = (unsigned*)(ad + (long)N * HEADS);
    float*    segsum = (float*)(segmax + (long)N * HEADS);
    float*    gate   = segsum + (long)N * HEADS;        // [N]
    float*    xg     = gate + N;                        // [256]
    float*    gav    = xg + OUTC;                       // [256]
    unsigned* gmax   = (unsigned*)(gav + OUTC);         // scalar
    float*    gsum   = (float*)(gmax + 1);              // scalar
    float*    y2     = xp;                              // reuse (xp dead post-scatter)
    float*    xfin   = (float*)d_out;                   // final features live in d_out

    const int B = 256;
    auto blocks = [](long n, int b) { return (unsigned)((n + b - 1) / b); };
    // wave-jobs for the 16x64-strip GEMM: tilesM * (Nc/64), 8 waves per block
    auto gemm_blocks = [](int M, int Nc) {
        long jobs = (long)((M + 15) / 16) * (Nc / 64);
        return (unsigned)((jobs + 7) / 8);
    };

    // 1) conversions for GEMM1
    k_f32_to_f16<<<blocks((long)N * 128, B), B, 0, stream>>>(x, h16, (long)N * 128);
    k_convert_W<<<blocks(HEADS * 128 * HD, B), B, 0, stream>>>(W, wf16);

    // 2) xp = x @ W  (WMMA, K=128)
    k_wmma_gemm<<<gemm_blocks(N, OUTC), B, 0, stream>>>(h16, wf16, nullptr, xp, N, 128, OUTC);

    // 3) attention coefficients
    k_alpha<<<blocks((long)N * HEADS, B), B, 0, stream>>>(xp, a_src, a_dst, as, ad, N);

    // 4) init aggregation state
    k_init_agg<<<blocks((long)N * OUTC, B), B, 0, stream>>>(agg, conv_b, (long)N * OUTC);
    k_init_seg<<<blocks((long)N * HEADS, B), B, 0, stream>>>(segmax, segsum, N * HEADS);

    // 5) segment softmax + message scatter (bandwidth-dominant phase)
    k_edge_max<<<blocks(Etot, B), B, 0, stream>>>(ei, E, N, as, ad, segmax);
    k_edge_sum<<<blocks(Etot, B), B, 0, stream>>>(ei, E, N, as, ad, segmax, segsum);
    k_edge_scatter<<<blocks(Etot * HEADS, B), B, 0, stream>>>(ei, E, N, as, ad, segmax,
                                                              segsum, xp, agg);

    // 6) dense re-attention: y1 = x_local @ fc_w^T + fc_b (WMMA, K=256)
    k_f32_to_f16<<<blocks((long)N * OUTC, B), B, 0, stream>>>(agg, h16, (long)N * OUTC);
    k_f32_to_f16<<<blocks(OUTC * OUTC, B), B, 0, stream>>>(fc_w, wf16, OUTC * OUTC);
    k_wmma_gemm<<<gemm_blocks(N, OUTC), B, 0, stream>>>(h16, wf16, fc_b, y1, N, OUTC, OUTC);

    // 7) x2 = leaky_relu(x_local * softmax(leaky_relu(y1)), 0.2) -> f16
    k_safuse<<<blocks(N, 8), B, 0, stream>>>(agg, y1, h16, N);

    // 8) y2 = x2 @ fc_w^T + fc_b (WMMA)
    k_wmma_gemm<<<gemm_blocks(N, OUTC), B, 0, stream>>>(h16, wf16, fc_b, y2, N, OUTC, OUTC);

    // 9) LayerNorm + L2 normalize + gate logits (writes every element of d_out)
    k_ln_l2_gate<<<blocks(N, 8), B, 0, stream>>>(y2, ln_w, ln_b, gate_w, gate_b, xfin, gate, N);

    // 10) global attention pooling (single graph)
    k_init_pool<<<1, 256, 0, stream>>>(xg, gmax, gsum);
    k_gate_max<<<blocks(N, B), B, 0, stream>>>(gate, gmax, N);
    k_gate_sum<<<blocks(N, B), B, 0, stream>>>(gate, gmax, gsum, N);
    k_pool<<<blocks((long)N * 32, B), B, 0, stream>>>(xfin, gate, gmax, gsum, xg, N);
    k_ga<<<1, 256, 0, stream>>>(xg, gfc_w, gfc_b, gav);

    // 11) out = x_final * ga (broadcast)
    k_final<<<blocks((long)N * OUTC, B), B, 0, stream>>>(xfin, gav, (long)N * OUTC);
}
